// Decoder_55173149884788
// MI455X (gfx1250) — matrix operand order repaired and checked by reading, not compiled, not run on hardware
//
#include <hip/hip_runtime.h>
#include <hip/hip_bf16.h>

// ---------------------------------------------------------------------------
// Transformer decoder layer for MI455X (gfx1250, wave32, WMMA bf16).
// GEMMs: D = A(M,K) x B(N,K)^T, f32 accumulation, v_wmma_f32_16x16x32_bf16.
// Data movement: async global->LDS (ASYNCcnt) double-buffered, K-step 64:
// 16 WMMAs per barrier pair, 8 async b128 in flight per thread.
// ---------------------------------------------------------------------------

typedef __bf16 bf16_t;
typedef __attribute__((ext_vector_type(16))) __bf16 v16bf;
typedef __attribute__((ext_vector_type(8)))  float  v8f;

union BFrag {
    uint4 u[2];
    v16bf v;
};

__device__ __forceinline__ bf16_t to_bf16(float f) { return (bf16_t)f; }

// Low 32 bits of a generic pointer to LDS == workgroup-relative LDS address.
__device__ __forceinline__ unsigned lds_addr32(const void* p) {
    return (unsigned)(unsigned long long)(uintptr_t)p;
}

// GLOBAL_LOAD_ASYNC_TO_LDS_B128: per-lane 16B global -> LDS, ASYNCcnt-tracked.
__device__ __forceinline__ void async_copy_b128(unsigned lds_off, const void* gptr) {
    asm volatile("global_load_async_to_lds_b128 %0, %1, off"
                 :: "v"(lds_off), "v"((unsigned long long)(uintptr_t)gptr)
                 : "memory");
}

__device__ __forceinline__ void wait_async_le8() {
    asm volatile("s_wait_asynccnt 0x8" ::: "memory");
}
__device__ __forceinline__ void wait_async_0() {
    asm volatile("s_wait_asynccnt 0x0" ::: "memory");
}

// ---------------------------------------------------------------------------
// f32 -> bf16 convert (weights, enc_output)
// ---------------------------------------------------------------------------
__global__ __launch_bounds__(256) void cvt_f32_bf16(const float* __restrict__ s,
                                                    bf16_t* __restrict__ d, int n) {
    int i = blockIdx.x * 256 + threadIdx.x;
    if (i < n) d[i] = to_bf16(s[i]);
}

// ---------------------------------------------------------------------------
// LayerNorm over last dim (Cdim), one block per row, bf16 output
// ---------------------------------------------------------------------------
__global__ __launch_bounds__(256) void layernorm_bf16(const float* __restrict__ x,
                                                      const float* __restrict__ g,
                                                      const float* __restrict__ b,
                                                      bf16_t* __restrict__ o, int Cdim) {
    __shared__ float red[256];
    const int tid = threadIdx.x;
    const float* xr = x + (size_t)blockIdx.x * Cdim;
    float s = 0.f, sq = 0.f;
    for (int i = tid; i < Cdim; i += 256) {
        float v = xr[i];
        s += v;
        sq += v * v;
    }
    red[tid] = s; __syncthreads();
    for (int st = 128; st > 0; st >>= 1) {
        if (tid < st) red[tid] += red[tid + st];
        __syncthreads();
    }
    float mean = red[0] / Cdim; __syncthreads();
    red[tid] = sq; __syncthreads();
    for (int st = 128; st > 0; st >>= 1) {
        if (tid < st) red[tid] += red[tid + st];
        __syncthreads();
    }
    float var = red[0] / Cdim - mean * mean;
    float inv = rsqrtf(var + 1e-5f);
    bf16_t* orow = o + (size_t)blockIdx.x * Cdim;
    for (int i = tid; i < Cdim; i += 256)
        orow[i] = to_bf16((xr[i] - mean) * inv * g[i] + b[i]);
}

// ---------------------------------------------------------------------------
// Row softmax in place (f32), one block per row
// ---------------------------------------------------------------------------
__global__ __launch_bounds__(256) void softmax_rows(float* __restrict__ p, int n) {
    __shared__ float red[256];
    const int tid = threadIdx.x;
    float* pr = p + (size_t)blockIdx.x * n;
    float mx = -3.4e38f;
    for (int i = tid; i < n; i += 256) mx = fmaxf(mx, pr[i]);
    red[tid] = mx; __syncthreads();
    for (int st = 128; st > 0; st >>= 1) {
        if (tid < st) red[tid] = fmaxf(red[tid], red[tid + st]);
        __syncthreads();
    }
    mx = red[0]; __syncthreads();
    float sum = 0.f;
    for (int i = tid; i < n; i += 256) {
        float e = __expf(pr[i] - mx);
        pr[i] = e;
        sum += e;
    }
    red[tid] = sum; __syncthreads();
    for (int st = 128; st > 0; st >>= 1) {
        if (tid < st) red[tid] += red[tid + st];
        __syncthreads();
    }
    float inv = 1.0f / red[0];
    for (int i = tid; i < n; i += 256) pr[i] *= inv;
}

// ---------------------------------------------------------------------------
// Per-head transpose via LDS tile (coalesced both sides):
// Vt[(b,h,d,s)] = V[(b,s, h*64+d)], 64x64 tiles.
// ---------------------------------------------------------------------------
__global__ __launch_bounds__(256) void transpose_heads(const bf16_t* __restrict__ V,
                                                       bf16_t* __restrict__ Vt,
                                                       int Sdim, int Cdim, int Hn) {
    __shared__ __align__(16) bf16_t tile[64][72];
    const int bh = blockIdx.y;
    const int b = bh / Hn, h = bh % Hn;
    const int s0 = blockIdx.x * 64;
    const int tid = threadIdx.x;
    // load 64 (s) x 64 (d): 512 chunks of 8 halves, coalesced b128 reads
    for (int i = 0; i < 2; ++i) {
        int idx = tid + i * 256;
        int r = idx >> 3, c = (idx & 7) * 8;
        *(uint4*)&tile[r][c] =
            *(const uint4*)(V + ((size_t)b * Sdim + s0 + r) * Cdim + h * 64 + c);
    }
    __syncthreads();
    // write 64 (d) x 64 (s): gather inside LDS, coalesced b128 writes
    for (int i = 0; i < 2; ++i) {
        int idx = tid + i * 256;
        int d = idx >> 3, c = (idx & 7) * 8;
        bf16_t tmp[8];
        for (int j = 0; j < 8; ++j) tmp[j] = tile[c + j][d];
        *(uint4*)(Vt + ((size_t)bh * 64 + d) * Sdim + s0 + c) = *(uint4*)tmp;
    }
}

// ---------------------------------------------------------------------------
// Generic WMMA GEMM: D(M,N) = A(M,K) x B(N,K)^T  (+bias)(+relu)(+residual)
// Block tile 128x128, 8 waves (4x2), each wave 32x64 (2x4 WMMA accumulators).
// K-step 64: double-buffered LDS (72KB) fed by async global->LDS b128.
// ---------------------------------------------------------------------------
__global__ __launch_bounds__(256) void gemm_bf16(const bf16_t* __restrict__ A,
                                                 const bf16_t* __restrict__ B,
                                                 const float* __restrict__ bias,
                                                 const float* __restrict__ resid,
                                                 float* __restrict__ outF,
                                                 bf16_t* __restrict__ outB,
                                                 int M, int N, int K, int relu) {
    __shared__ __align__(16) bf16_t As[2][128 * 72];
    __shared__ __align__(16) bf16_t Bs[2][128 * 72];
    const int tid = threadIdx.x;
    const int m0 = blockIdx.x * 128;
    const int n0 = blockIdx.y * 128;
    const int w = tid >> 5, lane = tid & 31;
    const int wm = (w & 3) * 32;   // 4 waves along M
    const int wn = (w >> 2) * 64;  // 2 waves along N
    const int lm = lane & 15, lh = lane >> 4;

    const bf16_t* Abase = A + (size_t)m0 * K;
    const bf16_t* Bbase = B + (size_t)n0 * K;

    // 128 rows x 64 halves per matrix = 1024 chunks of 8 halves -> 4/thread
    auto issue_tile = [&](int buf, int k0) {
        for (int i = 0; i < 4; ++i) {
            int idx = tid + i * 256;
            int r = idx >> 3, c = (idx & 7) * 8;
            async_copy_b128(lds_addr32(&As[buf][r * 72 + c]),
                            Abase + (size_t)r * K + k0 + c);
        }
        for (int i = 0; i < 4; ++i) {
            int idx = tid + i * 256;
            int r = idx >> 3, c = (idx & 7) * 8;
            async_copy_b128(lds_addr32(&Bs[buf][r * 72 + c]),
                            Bbase + (size_t)r * K + k0 + c);
        }
    };

    v8f acc[2][4] = {};
    const int nsteps = K >> 6;
    issue_tile(0, 0);
    for (int i = 0; i < nsteps; ++i) {
        const int cur = i & 1;
        if (i + 1 < nsteps) {
            issue_tile(cur ^ 1, (i + 1) << 6);  // prefetch next K-tile
            wait_async_le8();                   // oldest 8 (tile i) complete
        } else {
            wait_async_0();
        }
        __syncthreads();

        for (int ks = 0; ks < 2; ++ks) {
            const int ko = ks * 32;
            BFrag af[2], bf4[4];
            for (int t = 0; t < 2; ++t) {
                const bf16_t* ap = &As[cur][(wm + t * 16 + lm) * 72 + ko];
                af[t].u[0] = *(const uint4*)(ap + lh * 8);
                af[t].u[1] = *(const uint4*)(ap + 16 + lh * 8);
            }
            for (int t = 0; t < 4; ++t) {
                const bf16_t* bp = &Bs[cur][(wn + t * 16 + lm) * 72 + ko + lh * 8];
                bf4[t].u[0] = *(const uint4*)(bp);
                bf4[t].u[1] = *(const uint4*)(bp + 16);
            }
            for (int tm = 0; tm < 2; ++tm)
                for (int tn = 0; tn < 4; ++tn)
                    acc[tm][tn] = __builtin_amdgcn_wmma_f32_16x16x32_bf16(
                        false, af[tm].v, false, bf4[tn].v, (short)0, acc[tm][tn],
                        false, false);
        }
        __syncthreads();
    }

    for (int tm = 0; tm < 2; ++tm)
        for (int tn = 0; tn < 4; ++tn)
            for (int i = 0; i < 8; ++i) {
                int m = m0 + wm + tm * 16 + i + 8 * lh;
                int n = n0 + wn + tn * 16 + lm;
                float v = acc[tm][tn][i];
                if (bias) v += bias[n];
                if (relu) v = v > 0.f ? v : 0.f;
                size_t o = (size_t)m * N + n;
                if (resid) v += resid[o];
                if (outF) outF[o] = v;
                if (outB) outB[o] = to_bf16(v);
            }
}

// ---------------------------------------------------------------------------
// Attention scores: wei[b,h,t,s] = mask ? scale * Q_h K_h^T : -1e9
// K-dim = 64 (2 WMMA steps), fragments direct from global bf16.
// ---------------------------------------------------------------------------
__global__ __launch_bounds__(256) void attn_scores(const bf16_t* __restrict__ Q,
                                                   const bf16_t* __restrict__ Km,
                                                   const int* __restrict__ mask,
                                                   float* __restrict__ wei,
                                                   int T, int S, int Cdim, int Hn,
                                                   float scale) {
    const int bh = blockIdx.z;
    const int b = bh / Hn, h = bh % Hn;
    const int t0 = blockIdx.x * 64, s0 = blockIdx.y * 128;
    const bf16_t* Qb = Q + (size_t)b * T * Cdim + h * 64;
    const bf16_t* Kb = Km + (size_t)b * S * Cdim + h * 64;
    const int tid = threadIdx.x, w = tid >> 5, lane = tid & 31;
    const int wm = (w & 1) * 32, wn = (w >> 1) * 32;
    const int lm = lane & 15, lh = lane >> 4;

    v8f acc[2][2] = {};
    for (int k0 = 0; k0 < 64; k0 += 32) {
        BFrag af[2], bf[2];
        for (int t = 0; t < 2; ++t) {
            const bf16_t* ap = Qb + (size_t)(t0 + wm + t * 16 + lm) * Cdim + k0;
            af[t].u[0] = *(const uint4*)(ap + lh * 8);
            af[t].u[1] = *(const uint4*)(ap + 16 + lh * 8);
            const bf16_t* bp = Kb + (size_t)(s0 + wn + t * 16 + lm) * Cdim + k0 + lh * 8;
            bf[t].u[0] = *(const uint4*)(bp);
            bf[t].u[1] = *(const uint4*)(bp + 16);
        }
        for (int tm = 0; tm < 2; ++tm)
            for (int tn = 0; tn < 2; ++tn)
                acc[tm][tn] = __builtin_amdgcn_wmma_f32_16x16x32_bf16(
                    false, af[tm].v, false, bf[tn].v, (short)0, acc[tm][tn],
                    false, false);
    }
    float* wb = wei + (size_t)bh * T * S;
    const int* mb = mask + (size_t)b * T * S;
    for (int tm = 0; tm < 2; ++tm)
        for (int tn = 0; tn < 2; ++tn)
            for (int i = 0; i < 8; ++i) {
                int t = t0 + wm + tm * 16 + i + 8 * lh;
                int s = s0 + wn + tn * 16 + lm;
                float v = acc[tm][tn][i] * scale;
                size_t o = (size_t)t * S + s;
                if (mb[o] == 0) v = -1e9f;
                wb[o] = v;
            }
}

// ---------------------------------------------------------------------------
// P @ V per head: O[b,t, h*64+d] = sum_s P[b,h,t,s] * Vt[b,h,d,s]
// Block tile 128x64, 8 waves (4x2), A = f32 probs (float4 loads) -> bf16.
// ---------------------------------------------------------------------------
__global__ __launch_bounds__(256) void attn_pv(const float* __restrict__ P,
                                               const bf16_t* __restrict__ Vt,
                                               bf16_t* __restrict__ O,
                                               int T, int S, int Cdim, int Hn) {
    const int bh = blockIdx.y;
    const int b = bh / Hn, h = bh % Hn;
    const int t0 = blockIdx.x * 128;
    const float* Pb = P + (size_t)bh * T * S;
    const bf16_t* Vb = Vt + (size_t)bh * 64 * S;
    const int tid = threadIdx.x, w = tid >> 5, lane = tid & 31;
    const int wm = (w >> 1) * 32, wn = (w & 1) * 32;  // 4x2 wave grid
    const int lm = lane & 15, lh = lane >> 4;

    v8f acc[2][2] = {};
    for (int k0 = 0; k0 < S; k0 += 32) {
        BFrag af[2], bf[2];
        for (int t = 0; t < 2; ++t) {
            const float* ap = Pb + (size_t)(t0 + wm + t * 16 + lm) * S + k0;
            float4 p0 = *(const float4*)(ap + lh * 8);
            float4 p1 = *(const float4*)(ap + lh * 8 + 4);
            float4 p2 = *(const float4*)(ap + 16 + lh * 8);
            float4 p3 = *(const float4*)(ap + 16 + lh * 8 + 4);
            af[t].v[0] = to_bf16(p0.x);  af[t].v[1] = to_bf16(p0.y);
            af[t].v[2] = to_bf16(p0.z);  af[t].v[3] = to_bf16(p0.w);
            af[t].v[4] = to_bf16(p1.x);  af[t].v[5] = to_bf16(p1.y);
            af[t].v[6] = to_bf16(p1.z);  af[t].v[7] = to_bf16(p1.w);
            af[t].v[8] = to_bf16(p2.x);  af[t].v[9] = to_bf16(p2.y);
            af[t].v[10] = to_bf16(p2.z); af[t].v[11] = to_bf16(p2.w);
            af[t].v[12] = to_bf16(p3.x); af[t].v[13] = to_bf16(p3.y);
            af[t].v[14] = to_bf16(p3.z); af[t].v[15] = to_bf16(p3.w);
            const bf16_t* bp = Vb + (size_t)(wn + t * 16 + lm) * S + k0 + lh * 8;
            bf[t].u[0] = *(const uint4*)(bp);
            bf[t].u[1] = *(const uint4*)(bp + 16);
        }
        for (int tm = 0; tm < 2; ++tm)
            for (int tn = 0; tn < 2; ++tn)
                acc[tm][tn] = __builtin_amdgcn_wmma_f32_16x16x32_bf16(
                    false, af[tm].v, false, bf[tn].v, (short)0, acc[tm][tn],
                    false, false);
    }
    for (int tm = 0; tm < 2; ++tm)
        for (int tn = 0; tn < 2; ++tn)
            for (int i = 0; i < 8; ++i) {
                int t = t0 + wm + tm * 16 + i + 8 * lh;
                int d = wn + tn * 16 + lm;
                O[((size_t)b * T + t) * Cdim + h * 64 + d] = to_bf16(acc[tm][tn][i]);
            }
}

// ---------------------------------------------------------------------------
// Host-side orchestration
// ---------------------------------------------------------------------------
extern "C" void kernel_launch(void* const* d_in, const int* in_sizes, int n_in,
                              void* d_out, int out_size, void* d_ws, size_t ws_size,
                              hipStream_t stream) {
    (void)in_sizes; (void)n_in; (void)out_size; (void)ws_size;
    constexpr int Bq = 4, T = 1024, S = 1024, C = 1024, Hn = 16, FFd = 4096;
    constexpr int M = Bq * T;  // 4096 tokens

    const float* x        = (const float*)d_in[0];
    const float* enc      = (const float*)d_in[1];
    // d_in[2] = enc_mask (unused by reference)
    const int*   dec_mask = (const int*)d_in[3];
    const int*   c_mask   = (const int*)d_in[4];
    const float* wq1 = (const float*)d_in[5];
    const float* wk1 = (const float*)d_in[6];
    const float* wv1 = (const float*)d_in[7];
    const float* wo1 = (const float*)d_in[8];
    const float* wq2 = (const float*)d_in[9];
    const float* wk2 = (const float*)d_in[10];
    const float* wv2 = (const float*)d_in[11];
    const float* wo2 = (const float*)d_in[12];
    const float* ln1_g = (const float*)d_in[13];
    const float* ln1_b = (const float*)d_in[14];
    const float* ln2_g = (const float*)d_in[15];
    const float* ln2_b = (const float*)d_in[16];
    const float* ln3_g = (const float*)d_in[17];
    const float* ln3_b = (const float*)d_in[18];
    const float* w_ff1 = (const float*)d_in[19];
    const float* b_ff1 = (const float*)d_in[20];
    const float* w_ff2 = (const float*)d_in[21];
    const float* b_ff2 = (const float*)d_in[22];

    float* xout = (float*)d_out;                       // (B,T,C) f32
    float* wei  = (float*)d_out + (size_t)M * C;       // (B,H,T,S) f32 scratch + output

    // Workspace layout (bytes)
    char* ws = (char*)d_ws;
    const size_t MB = 1024 * 1024;
    bf16_t* wb   = (bf16_t*)(ws + 0);        // 16M bf16 = 32MB weights
    float*  res  = (float*) (ws + 32 * MB);  // 16MB residual accumulator
    bf16_t* lnb  = (bf16_t*)(ws + 48 * MB);  // 8MB  LN output
    bf16_t* encb = (bf16_t*)(ws + 56 * MB);  // 8MB  enc_output bf16
    bf16_t* qb   = (bf16_t*)(ws + 64 * MB);  // 8MB
    bf16_t* kb   = (bf16_t*)(ws + 72 * MB);  // 8MB
    bf16_t* vb   = (bf16_t*)(ws + 80 * MB);  // 8MB
    bf16_t* vtb  = (bf16_t*)(ws + 88 * MB);  // 8MB per-head-transposed V
    bf16_t* ffh  = qb;                       // 32MB alias over qb..vtb (FF phase only)
    bf16_t* aout = (bf16_t*)(ws + 96 * MB);  // 8MB attention output (B,T,C)

    const int CC = C * C;
    bf16_t* wq1b = wb + (size_t)0 * CC;
    bf16_t* wk1b = wb + (size_t)1 * CC;
    bf16_t* wv1b = wb + (size_t)2 * CC;
    bf16_t* wo1b = wb + (size_t)3 * CC;
    bf16_t* wq2b = wb + (size_t)4 * CC;
    bf16_t* wk2b = wb + (size_t)5 * CC;
    bf16_t* wv2b = wb + (size_t)6 * CC;
    bf16_t* wo2b = wb + (size_t)7 * CC;
    bf16_t* wf1b = wb + (size_t)8 * CC;      // FFd*C = 4M
    bf16_t* wf2b = wb + (size_t)12 * CC;     // C*FFd = 4M

    // --- weight / enc conversion to bf16 ---
    cvt_f32_bf16<<<CC / 256, 256, 0, stream>>>(wq1, wq1b, CC);
    cvt_f32_bf16<<<CC / 256, 256, 0, stream>>>(wk1, wk1b, CC);
    cvt_f32_bf16<<<CC / 256, 256, 0, stream>>>(wv1, wv1b, CC);
    cvt_f32_bf16<<<CC / 256, 256, 0, stream>>>(wo1, wo1b, CC);
    cvt_f32_bf16<<<CC / 256, 256, 0, stream>>>(wq2, wq2b, CC);
    cvt_f32_bf16<<<CC / 256, 256, 0, stream>>>(wk2, wk2b, CC);
    cvt_f32_bf16<<<CC / 256, 256, 0, stream>>>(wv2, wv2b, CC);
    cvt_f32_bf16<<<CC / 256, 256, 0, stream>>>(wo2, wo2b, CC);
    cvt_f32_bf16<<<(FFd * C) / 256, 256, 0, stream>>>(w_ff1, wf1b, FFd * C);
    cvt_f32_bf16<<<(C * FFd) / 256, 256, 0, stream>>>(w_ff2, wf2b, C * FFd);
    cvt_f32_bf16<<<(Bq * S * C) / 256, 256, 0, stream>>>(enc, encb, Bq * S * C);

    // res = x
    hipMemcpyAsync(res, x, (size_t)M * C * sizeof(float),
                   hipMemcpyDeviceToDevice, stream);

    const dim3 blk(256);
    const dim3 gP(M / 128, C / 128);      // projection GEMM grid
    const dim3 gSc(T / 64, S / 128, Bq * Hn);
    const dim3 gPV(T / 128, Bq * Hn);
    const dim3 gTr(S / 64, Bq * Hn);
    const float scale = 0.125f;           // (C/H)^-0.5 = 1/8

    // ================= self-attention =================
    layernorm_bf16<<<M, blk, 0, stream>>>(res, ln1_g, ln1_b, lnb, C);
    gemm_bf16<<<gP, blk, 0, stream>>>(lnb, wq1b, nullptr, nullptr, nullptr, qb, M, C, C, 0);
    gemm_bf16<<<gP, blk, 0, stream>>>(lnb, wk1b, nullptr, nullptr, nullptr, kb, M, C, C, 0);
    gemm_bf16<<<gP, blk, 0, stream>>>(lnb, wv1b, nullptr, nullptr, nullptr, vb, M, C, C, 0);
    attn_scores<<<gSc, blk, 0, stream>>>(qb, kb, dec_mask, wei, T, S, C, Hn, scale);
    softmax_rows<<<Bq * Hn * T, blk, 0, stream>>>(wei, S);
    transpose_heads<<<gTr, blk, 0, stream>>>(vb, vtb, S, C, Hn);
    attn_pv<<<gPV, blk, 0, stream>>>(wei, vtb, aout, T, S, C, Hn);
    gemm_bf16<<<gP, blk, 0, stream>>>(aout, wo1b, nullptr, res, res, nullptr, M, C, C, 0);

    // ================= cross-attention =================
    layernorm_bf16<<<M, blk, 0, stream>>>(res, ln2_g, ln2_b, lnb, C);
    gemm_bf16<<<gP, blk, 0, stream>>>(lnb,  wq2b, nullptr, nullptr, nullptr, qb, M, C, C, 0);
    gemm_bf16<<<gP, blk, 0, stream>>>(encb, wk2b, nullptr, nullptr, nullptr, kb, M, C, C, 0);
    gemm_bf16<<<gP, blk, 0, stream>>>(encb, wv2b, nullptr, nullptr, nullptr, vb, M, C, C, 0);
    attn_scores<<<gSc, blk, 0, stream>>>(qb, kb, c_mask, wei, T, S, C, Hn, scale);
    softmax_rows<<<Bq * Hn * T, blk, 0, stream>>>(wei, S);  // final `wei` output
    transpose_heads<<<gTr, blk, 0, stream>>>(vb, vtb, S, C, Hn);
    attn_pv<<<gPV, blk, 0, stream>>>(wei, vtb, aout, T, S, C, Hn);
    gemm_bf16<<<gP, blk, 0, stream>>>(aout, wo2b, nullptr, res, res, nullptr, M, C, C, 0);

    // ================= feed-forward =================
    layernorm_bf16<<<M, blk, 0, stream>>>(res, ln3_g, ln3_b, lnb, C);
    gemm_bf16<<<dim3(M / 128, FFd / 128), blk, 0, stream>>>(
        lnb, wf1b, b_ff1, nullptr, nullptr, ffh, M, FFd, C, 1);
    gemm_bf16<<<dim3(M / 128, C / 128), blk, 0, stream>>>(
        ffh, wf2b, b_ff2, res, xout, nullptr, M, C, FFd, 0);
}
